// EQDockModel_90563680403841
// MI455X (gfx1250) — matrix operand order repaired and checked by reading, not compiled, not run on hardware
//
#include <hip/hip_runtime.h>

typedef unsigned short u16;
typedef unsigned int   u32;
typedef unsigned long long u64;
typedef __attribute__((ext_vector_type(8)))  u16    u16x8;
typedef __attribute__((ext_vector_type(16))) u16    u16x16;
typedef __attribute__((ext_vector_type(16))) __bf16 v16bf;
typedef __attribute__((ext_vector_type(8)))  float  v8f;
typedef __attribute__((ext_vector_type(4)))  u32    u32x4;
typedef __attribute__((ext_vector_type(8)))  int    i32x8;
typedef __attribute__((ext_vector_type(4)))  int    i32x4;

#define NB    16
#define NANG  120
#define APAD  128
#define HH    50
#define SS    100
#define ROWW  64          // j padded 50 -> 64 so K-chunks never straddle rows
#define KTOT  6400        // (2*50) rows * 64
#define PADH  152         // receptor pad rows (need 149)
#define PADW  164         // receptor pad cols (need 163)
#define NPOS  10000
#define NTILE 64          // 4 WMMA n-tiles per workgroup
#define LDSELEM (PADH * PADW)   // 24928 bf16 = 49856 B

// ---------- helpers ----------
__device__ inline u16 f2bf(float f) {
    u32 u = __float_as_uint(f);
    u32 r = u + 0x7FFFu + ((u >> 16) & 1u);          // RNE
    if ((u & 0x7F800000u) == 0x7F800000u) r = u;     // inf/nan passthrough
    return (u16)(r >> 16);
}
__device__ inline u64 mkkey(float v, u32 idx) {
    u32 u = __float_as_uint(v);
    u32 s = (u & 0x80000000u) ? ~u : (u | 0x80000000u); // sortable float
    return ((u64)s << 32) | idx;
}

// ---------- 0: init argmin keys ----------
__global__ void k_init(u64* keys) {
    int t = threadIdx.x;
    if (t < 32) keys[t] = ~0ull;
}

// ---------- 1: conv3x3 + bias + relu (SAME) ----------
__global__ void k_conv(const float* __restrict__ x, const float* __restrict__ w,
                       const float* __restrict__ bias, float* __restrict__ y) {
    int tid = blockIdx.x * 256 + threadIdx.x;
    if (tid >= NB * 2 * HH * HH) return;
    int p = tid % (HH * HH);
    int o = (tid / (HH * HH)) & 1;
    int b = tid / (2 * HH * HH);
    int i = p / HH, j = p % HH;
    const float* xb = x + (size_t)b * HH * HH;   // single input channel
    float s = bias[o];
#pragma unroll
    for (int di = 0; di < 3; ++di) {
        int yy = i + di - 1; if (yy < 0 || yy >= HH) continue;
#pragma unroll
        for (int dj = 0; dj < 3; ++dj) {
            int xx = j + dj - 1; if (xx < 0 || xx >= HH) continue;
            s += xb[yy * HH + xx] * w[o * 9 + di * 3 + dj];
        }
    }
    y[tid] = s > 0.f ? s : 0.f;
}

// ---------- 2: zero-padded bf16 receptor image for shifted gathers ----------
__global__ void k_recp(const float* __restrict__ rec_feat, u16* __restrict__ recp) {
    int tid = blockIdx.x * 256 + threadIdx.x;
    if (tid >= NB * 2 * PADH * PADW) return;
    int s = tid % PADW;
    int r = (tid / PADW) % PADH;
    int c = (tid / (PADW * PADH)) % 2;
    int b = tid / (2 * PADW * PADH);
    float v = 0.f;
    if (r >= 49 && r < 49 + HH && s >= 49 && s < 49 + HH)
        v = rec_feat[(((size_t)b * 2 + c) * HH + (r - 49)) * HH + (s - 49)];
    recp[tid] = f2bf(v);
}

// ---------- 3: rotate ligand (bilinear) + fold w_score -> bf16 A matrix ----------
__global__ void k_wlig(const float* __restrict__ lig_feat, const float* __restrict__ wsc,
                       u16* __restrict__ wlig) {
    int tid = blockIdx.x * 256 + threadIdx.x;          // NB*APAD*50*64 threads
    int r = tid % (HH * ROWW);
    int a = (tid / (HH * ROWW)) % APAD;
    int b = tid / (APAD * HH * ROWW);
    int i = r >> 6, j = r & 63;
    size_t obase = ((size_t)(b * APAD + a)) * KTOT;
    u16* o0 = wlig + obase + (size_t)i * ROWW + j;          // c=0
    u16* o1 = wlig + obase + (size_t)(HH + i) * ROWW + j;   // c=1
    if (a >= NANG || j >= HH) { *o0 = 0; *o1 = 0; return; }
    float ang = 0.05235987755982988f * (float)a;   // 2*pi/120
    float ca = cosf(ang), sa = sinf(ang);
    float gx = -1.f + (float)j * (2.f / 49.f);
    float gy = -1.f + (float)i * (2.f / 49.f);
    float xin = ca * gx - sa * gy;
    float yin = sa * gx + ca * gy;
    float px = (xin + 1.f) * 24.5f, py = (yin + 1.f) * 24.5f;
    int x0 = (int)floorf(px), y0 = (int)floorf(py);
    float wx1 = px - (float)x0, wx0 = 1.f - wx1;
    float wy1 = py - (float)y0, wy0 = 1.f - wy1;
    const float* f0 = lig_feat + (size_t)(b * 2 + 0) * HH * HH;
    const float* f1 = lig_feat + (size_t)(b * 2 + 1) * HH * HH;
    float s0 = 0.f, s1 = 0.f;
#pragma unroll
    for (int t = 0; t < 4; ++t) {
        int yi = y0 + (t >> 1), xi = x0 + (t & 1);
        float wgt = (t >> 1 ? wy1 : wy0) * (t & 1 ? wx1 : wx0);
        if (yi >= 0 && yi < HH && xi >= 0 && xi < HH) {
            s0 += f0[yi * HH + xi] * wgt;
            s1 += f1[yi * HH + xi] * wgt;
        }
    }
    *o0 = f2bf(wsc[0] * s0 + wsc[1] * s1);
    *o1 = f2bf(wsc[2] * s0 + wsc[3] * s1);
}

// ---------- 4: WMMA GEMM (M=128 angles, N=64 positions/block, K=6400) + fused argmin ----------
__global__ void __launch_bounds__(256)
k_gemm(const u16* __restrict__ wlig, const u16* __restrict__ recp,
       const float* __restrict__ bsc, u64* __restrict__ keys) {
    __shared__ u16 sm[LDSELEM];                 // one channel of padded receptor: ~49KB

    const int b    = blockIdx.y;
    const int base = blockIdx.x * NTILE;
    const int tid  = threadIdx.x;
    const int lane = tid & 31, wave = tid >> 5;
    const int nl = lane & 15, kh = lane >> 4;

    int  posv[4]; bool vv[4]; int ro[4], co[4];
#pragma unroll
    for (int t = 0; t < 4; ++t) {
        posv[t] = base + nl + 16 * t;
        vv[t]   = posv[t] < NPOS;
        ro[t]   = vv[t] ? (99 - posv[t] / SS) : 0;    // 49 - dx
        co[t]   = vv[t] ? (99 - posv[t] % SS) : 0;    // 49 - dy
    }

    // A operand: row M = lane%16 of this wave's angle tile; k-half by lane/16
    const u16* ap = wlig + ((size_t)(b * APAD + wave * 16 + nl)) * KTOT + kh * 16;

    v8f acc[4];
#pragma unroll
    for (int t = 0; t < 4; ++t) acc[t] = (v8f){};

    for (int ph = 0; ph < 2; ++ph) {                // channel c = ph
        const u16* gsrc = recp + ((size_t)(b * 2 + ph)) * LDSELEM;
#if __has_builtin(__builtin_amdgcn_tensor_load_to_lds)
        // Tensor Data Mover: DMA the padded receptor channel into LDS (1D tile).
        if (wave == 0) {
            u64 gaddr = (u64)(uintptr_t)gsrc;
            u32 ldsa  = (u32)(uintptr_t)(&sm[0]);
            u32x4 g0;
            g0[0] = 1u;                                         // count=1 (user, valid)
            g0[1] = ldsa;                                       // lds_addr
            g0[2] = (u32)gaddr;                                 // global_addr[31:0]
            g0[3] = (u32)((gaddr >> 32) & 0x01FFFFFFu)          // global_addr[56:32]
                  | (2u << 30);                                 // type = 2 (image)
            i32x8 g1;
            g1[0] = (int)(1u << 16);                            // data_size = 2B; no mask/flags
            g1[1] = (int)((LDSELEM & 0xFFFFu) << 16);           // tensor_dim0[15:0]
            g1[2] = (int)(1u << 16);                            // tensor_dim0[31:16]=0, tensor_dim1=1
            g1[3] = (int)((LDSELEM & 0xFFFFu) << 16);           // tensor_dim1 hi=0, tile_dim0=LDSELEM
            g1[4] = 1;                                          // tile_dim1=1, tile_dim2=0
            g1[5] = LDSELEM;                                    // tensor_dim0_stride[31:0]
            g1[6] = 0;                                          // stride hi / tensor_dim1_stride lo
            g1[7] = 0;
            i32x4 z4 = {0, 0, 0, 0};                            // D# group 2 (unused, <=2D)
            i32x8 z8 = {0, 0, 0, 0, 0, 0, 0, 0};                // D# group 3 / extra (unused)
            __builtin_amdgcn_tensor_load_to_lds(g0, g1, z4, z4, z8, 0);
            __builtin_amdgcn_s_wait_tensorcnt((short)0);
        }
#else
        {
            const u16x8* src = (const u16x8*)gsrc;
            u16x8* dst = (u16x8*)sm;
            for (int idx = tid; idx < LDSELEM / 8; idx += 256) dst[idx] = src[idx];
        }
#endif
        __syncthreads();

        for (int kb = ph * 100; kb < ph * 100 + 100; ++kb) {
            const u16* a0 = ap + kb * 32;
            if ((kb & 7) == 0) __builtin_prefetch(a0 + 256, 0, 1);  // ~8 k-steps ahead
            u16x8 alo = *(const u16x8*)a0;                          // K = kb*32 + kh*16 + e
            u16x8 ahi = *(const u16x8*)(a0 + 8);                    // K = +8
            u16x16 at = __builtin_shufflevector(alo, ahi,
                0,1,2,3,4,5,6,7,8,9,10,11,12,13,14,15);
            v16bf av = __builtin_bit_cast(v16bf, at);

            const int i  = (kb >> 1) - ph * HH;                     // image row of this k-chunk
            const int j0 = ((kb & 1) << 5) + (kh << 4);             // col base within row
#pragma unroll
            for (int t = 0; t < 4; ++t) {
                const u16* rp = sm + (i + ro[t]) * PADW + (j0 + co[t]);
                u16 tb[16];
                __builtin_memcpy(tb, rp, 32);                       // 16 contiguous bf16
                u16x16 bt;
#pragma unroll
                for (int e = 0; e < 16; ++e) bt[e] = tb[e];
                v16bf bv = __builtin_bit_cast(v16bf, bt);
                acc[t] = __builtin_amdgcn_wmma_f32_16x16x32_bf16(
                    false, av, false, bv, (short)0, acc[t], false, false);
            }
        }
        __syncthreads();
    }

    // fused argmin epilogue: C layout row m = r + 8*kh, col n = lane%16
    const float bias = bsc[0];
    u64 best = ~0ull, best0 = ~0ull;
#pragma unroll
    for (int r = 0; r < 8; ++r) {
        int m = r + kh * 8;
        int aidx = wave * 16 + m;
        if (aidx < NANG) {
#pragma unroll
            for (int t = 0; t < 4; ++t) {
                if (vv[t]) {
                    u64 k = mkkey(acc[t][r] + bias, (u32)(aidx * NPOS + posv[t]));
                    best = k < best ? k : best;
                    if (aidx == 0) {
                        u64 k0 = mkkey(acc[t][r] + bias, (u32)posv[t]);
                        best0 = k0 < best0 ? k0 : best0;
                    }
                }
            }
        }
    }
#pragma unroll
    for (int off = 16; off; off >>= 1) {
        u64 o = __shfl_xor(best, off, 32);  best  = o < best  ? o : best;
        u64 p = __shfl_xor(best0, off, 32); best0 = p < best0 ? p : best0;
    }
    if (lane == 0) {
        atomicMin(&keys[b], best);
        atomicMin(&keys[16 + b], best0);
    }
}

// ---------- 5: decode winner -> (angle, translation) ----------
__global__ void k_final(const u64* __restrict__ keys, float* __restrict__ out) {
    int b = threadIdx.x;
    if (b >= NB) return;
    u64 k = keys[b];
    u32 s = (u32)(k >> 32);
    u32 u = (s >> 31) ? (s & 0x7FFFFFFFu) : ~s;
    float val = __uint_as_float(u);
    int aidx, pos;
    if (val > 0.f) {
        // reference greedy's zero-refill: all-positive scores -> angle 0, slice-0 argmin
        aidx = 0;
        pos = (int)(keys[16 + b] & 0xFFFFFFFFull);
    } else {
        u32 idx = (u32)(k & 0xFFFFFFFFull);
        aidx = (int)(idx / NPOS);
        pos  = (int)(idx % NPOS);
    }
    out[b] = (float)aidx * 0.05235987755982988f;           // ang[a]
    out[16 + 2 * b + 0] = (float)(pos / SS) - 50.f;        // x - S/2
    out[16 + 2 * b + 1] = (float)(pos % SS) - 50.f;        // y - S/2
}

// ---------- host ----------
extern "C" void kernel_launch(void* const* d_in, const int* in_sizes, int n_in,
                              void* d_out, int out_size, void* d_ws, size_t ws_size,
                              hipStream_t stream) {
    (void)in_sizes; (void)n_in; (void)out_size; (void)ws_size;
    const float* receptor = (const float*)d_in[0];
    const float* ligand   = (const float*)d_in[1];
    const float* w_repr   = (const float*)d_in[2];
    const float* b_repr   = (const float*)d_in[3];
    const float* w_score  = (const float*)d_in[4];
    const float* b_score  = (const float*)d_in[5];

    char* ws = (char*)d_ws;
    // workspace layout (256B aligned)
    float* rec_feat = (float*)(ws + 0);                          // 16*2*2500 f32 = 320000B
    float* lig_feat = (float*)(ws + 320000);                     // 320000B
    u16*   recp     = (u16*)  (ws + 640000);                     // 16*2*152*164 bf16 = 1595392B
    u16*   wlig     = (u16*)  (ws + 2235392);                    // 16*128*6400 bf16 = 26214400B
    u64*   keys     = (u64*)  (ws + 28449792);                   // 32 u64

    k_init<<<1, 32, 0, stream>>>(keys);
    k_conv<<<313, 256, 0, stream>>>(receptor, w_repr, b_repr, rec_feat);
    k_conv<<<313, 256, 0, stream>>>(ligand,   w_repr, b_repr, lig_feat);
    k_recp<<<(NB * 2 * PADH * PADW) / 256, 256, 0, stream>>>(rec_feat, recp);
    k_wlig<<<(NB * APAD * HH * ROWW) / 256, 256, 0, stream>>>(lig_feat, w_score, wlig);
    k_gemm<<<dim3((NPOS + NTILE - 1) / NTILE, NB), 256, 0, stream>>>(wlig, recp, b_score, keys);
    k_final<<<1, 16, 0, stream>>>(keys, (float*)d_out);
}
